// MultiHeadSelfAttentionModel_69123203661968
// MI455X (gfx1250) — matrix operand order repaired and checked by reading, not compiled, not run on hardware
//
#include <hip/hip_runtime.h>
#include <hip/hip_bf16.h>

typedef __attribute__((ext_vector_type(16))) _Float16 v16h;
typedef __attribute__((ext_vector_type(8)))  float    v8f;

#define B_   8
#define S_   2048
#define D_   256
#define H_   4
#define HD_  64
#define BS_  (B_ * S_)          // 16384
#define EPS_ 1e-5f

// ---------------------------------------------------------------------------
// Prep kernels
// ---------------------------------------------------------------------------
__global__ void convert_w_kernel(const float* __restrict__ Wq,
                                 const float* __restrict__ Wk,
                                 const float* __restrict__ Wv,
                                 const float* __restrict__ Wo,
                                 _Float16* __restrict__ W16) {
  int i = blockIdx.x * blockDim.x + threadIdx.x;   // < 4*65536
  int wsel = i >> 16;
  int off  = i & 65535;
  const float* s = (wsel == 0) ? Wq : (wsel == 1) ? Wk : (wsel == 2) ? Wv : Wo;
  W16[i] = (_Float16)s[off];
}

__global__ void gather_kernel(const int* __restrict__ ids,
                              const float* __restrict__ emb,
                              _Float16* __restrict__ E16) {
  int i = blockIdx.x * blockDim.x + threadIdx.x;   // < BS_*D_
  int row = i >> 8;
  int col = i & 255;
  int tok = ids[row];
  E16[i] = (_Float16)emb[(size_t)tok * D_ + col];
}

__global__ __launch_bounds__(256)
void masksum_kernel(const int* __restrict__ mask, float* __restrict__ msum) {
  __shared__ float red[256];
  int b = blockIdx.x, t = threadIdx.x;
  float s = 0.0f;
  for (int i = t; i < S_; i += 256) s += (float)mask[b * S_ + i];
  red[t] = s; __syncthreads();
  for (int k = 128; k > 0; k >>= 1) { if (t < k) red[t] += red[t + k]; __syncthreads(); }
  if (t == 0) msum[b] = red[0];
}

// ---------------------------------------------------------------------------
// Tiled WMMA GEMM: C[BS x 256] = A[BS x 256] * W[256 x 256]
// 128 threads = 4 waves; block tile 64 rows x 64 cols; wave = 64x16 strip.
// MODE 0: f16 out = acc * scale        (QKV projections)
// MODE 1: f32 out = acc + bias + emb   (Wo projection + residual)
// ---------------------------------------------------------------------------
template <int MODE>
__global__ __launch_bounds__(128)
void gemm128(const _Float16* __restrict__ A,
             const _Float16* __restrict__ W,
             _Float16* __restrict__ outH,
             float* __restrict__ outF,
             float scale,
             const int* __restrict__ ids,
             const float* __restrict__ emb,
             const float* __restrict__ bias) {
  __shared__ _Float16 At[64 * 32];
  __shared__ _Float16 Bt[32 * 64];

  const int rowBase = blockIdx.x * 64;
  const int n0      = blockIdx.y * 64;
  const int t    = threadIdx.x;
  const int wave = t >> 5;
  const int lane = t & 31;
  const int g    = lane >> 4;
  const int m    = lane & 15;

  v8f acc[4] = {};

  for (int k0 = 0; k0 < D_; k0 += 32) {
    { // stage A tile 64x32 (each thread: 16 contiguous halves = 32 B)
      int r = t >> 1, off = (t & 1) * 16;
      const float4* src = (const float4*)(A + (size_t)(rowBase + r) * D_ + k0 + off);
      float4* dst = (float4*)(At + r * 32 + off);
      dst[0] = src[0]; dst[1] = src[1];
      if (k0 + 32 < D_)  // prefetch next A tile chunk into cache hierarchy
        __builtin_prefetch(A + (size_t)(rowBase + r) * D_ + k0 + 32 + off, 0, 0);
    }
    { // stage B tile 32x64
      int kk = t >> 2, off = (t & 3) * 16;
      const float4* src = (const float4*)(W + (size_t)(k0 + kk) * D_ + n0 + off);
      float4* dst = (float4*)(Bt + kk * 64 + off);
      dst[0] = src[0]; dst[1] = src[1];
      if (k0 + 32 < D_)
        __builtin_prefetch(W + (size_t)(k0 + 32 + kk) * D_ + n0 + off, 0, 0);
    }
    __syncthreads();

    v16h bf;
#pragma unroll
    for (int h = 0; h < 16; ++h) {
      int kk = h + 16 * g;                         // B: K = half + 16*(lane>=16)
      bf[h] = Bt[kk * 64 + wave * 16 + m];
    }
#pragma unroll
    for (int mi = 0; mi < 4; ++mi) {
      v16h af;
#pragma unroll
      for (int h = 0; h < 16; ++h) {
        int kk = h + 16 * g;   // A K-mapping
        af[h] = At[(mi * 16 + m) * 32 + kk];
      }
      acc[mi] = __builtin_amdgcn_wmma_f32_16x16x32_f16(
          false, af, false, bf, (short)0, acc[mi], false, false);
    }
    __syncthreads();
  }

#pragma unroll
  for (int mi = 0; mi < 4; ++mi)
#pragma unroll
    for (int i = 0; i < 8; ++i) {
      int row = rowBase + mi * 16 + i + 8 * g;     // C: row = i + 8*(lane>=16)
      int col = n0 + wave * 16 + m;                // C: col = lane & 15
      float v = acc[mi][i];
      if (MODE == 0) {
        outH[(size_t)row * D_ + col] = (_Float16)(v * scale);
      } else {
        int tok = ids[row];
        outF[(size_t)row * D_ + col] = v + bias[col] + emb[(size_t)tok * D_ + col];
      }
    }
}

// ---------------------------------------------------------------------------
// Flash attention: one wave per (b, head, 16-query tile). 32 keys per step.
// K/V tiles staged memory->LDS with CDNA5 async-LDS loads (ASYNCcnt).
// Q is pre-scaled by 1/sqrt(HD) at projection time.
// ---------------------------------------------------------------------------
__global__ __launch_bounds__(32)
void attention_kernel(const _Float16* __restrict__ Q,
                      const _Float16* __restrict__ K,
                      const _Float16* __restrict__ V,
                      const int* __restrict__ mask,
                      _Float16* __restrict__ Att) {
  __shared__ _Float16 kT[32 * 64];
  __shared__ _Float16 vT[32 * 64];
  __shared__ _Float16 pT[16 * 32];

  const int q0 = blockIdx.x * 16;
  const int hh = blockIdx.y;
  const int b  = blockIdx.z;
  const int lane = threadIdx.x;
  const int g = lane >> 4;
  const int m = lane & 15;

  // Q as two 16x32 A-fragments over d = 0..31 / 32..63
  v16h qa0, qa1;
  const _Float16* qrow = Q + ((size_t)(b * S_) + q0 + m) * D_ + hh * HD_;
#pragma unroll
  for (int h = 0; h < 16; ++h) {
    int kk = h + 16 * g;
    qa0[h] = qrow[kk];
    qa1[h] = qrow[kk + 32];
  }

  float mrow[8], lrow[8];
#pragma unroll
  for (int i = 0; i < 8; ++i) { mrow[i] = -1e30f; lrow[i] = 0.0f; }
  v8f acc[4] = {};

  // per-lane LDS byte addresses for async staging (lane owns one K/V row)
  const unsigned ldsK = (unsigned)(uintptr_t)kT + (unsigned)(lane * 128);
  const unsigned ldsV = (unsigned)(uintptr_t)vT + (unsigned)(lane * 128);
  const unsigned voff = (unsigned)(lane * (D_ * 2));   // row stride in bytes

  for (int k0 = 0; k0 < S_; k0 += 32) {
    { // async stage: lane l copies key/value row k0+l (128 B) memory -> LDS
      const _Float16* kbase = K + ((size_t)(b * S_) + k0) * D_ + hh * HD_;
      const _Float16* vbase = V + ((size_t)(b * S_) + k0) * D_ + hh * HD_;
#pragma unroll
      for (int j = 0; j < 8; ++j) {
        asm volatile("global_load_async_to_lds_b128 %0, %1, %2 offset:%3"
                     :: "v"(ldsK), "v"(voff), "s"(kbase), "i"(j * 16) : "memory");
        asm volatile("global_load_async_to_lds_b128 %0, %1, %2 offset:%3"
                     :: "v"(ldsV), "v"(voff), "s"(vbase), "i"(j * 16) : "memory");
      }
      asm volatile("s_wait_asynccnt 0x0" ::: "memory");
    }
    __syncthreads();

    // scores for 2 column tiles of 16 keys each
    v8f sc[2];
#pragma unroll
    for (int ct = 0; ct < 2; ++ct) {
      v16h bf0, bf1;
#pragma unroll
      for (int h = 0; h < 16; ++h) {
        int d = h + 16 * g;
        bf0[h] = kT[(ct * 16 + m) * 64 + d];        // B(d, n) = K[key=n, d]
        bf1[h] = kT[(ct * 16 + m) * 64 + d + 32];
      }
      v8f s = {};
      s = __builtin_amdgcn_wmma_f32_16x16x32_f16(false, qa0, false, bf0, (short)0, s, false, false);
      s = __builtin_amdgcn_wmma_f32_16x16x32_f16(false, qa1, false, bf1, (short)0, s, false, false);
      int mk = mask[b * S_ + k0 + ct * 16 + m];
      if (mk == 0) {
#pragma unroll
        for (int i = 0; i < 8; ++i) s[i] = -1e30f;
      }
      sc[ct] = s;
    }

    // online softmax per owned row slot (rows i + 8*g, reduce over 16 key-lanes)
#pragma unroll
    for (int i = 0; i < 8; ++i) {
      float tmax = fmaxf(sc[0][i], sc[1][i]);
#pragma unroll
      for (int off = 1; off < 16; off <<= 1) tmax = fmaxf(tmax, __shfl_xor(tmax, off, 32));
      float mnew = fmaxf(mrow[i], tmax);
      float resc = __expf(mrow[i] - mnew);
      float p0 = __expf(sc[0][i] - mnew);
      float p1 = __expf(sc[1][i] - mnew);
      float rsum = p0 + p1;
#pragma unroll
      for (int off = 1; off < 16; off <<= 1) rsum += __shfl_xor(rsum, off, 32);
      lrow[i] = lrow[i] * resc + rsum;
      mrow[i] = mnew;
#pragma unroll
      for (int nt = 0; nt < 4; ++nt) acc[nt][i] *= resc;
      pT[(i + 8 * g) * 32 + m]      = (_Float16)p0;  // transpose P through LDS
      pT[(i + 8 * g) * 32 + 16 + m] = (_Float16)p1;
    }
    __syncthreads();

    // attended += P(16x32) x V(32x64): 1 A-frag, 4 B-frags
    v16h pa;
#pragma unroll
    for (int h = 0; h < 16; ++h) {
      int kk = h + 16 * g;
      pa[h] = pT[m * 32 + kk];
    }
#pragma unroll
    for (int nt = 0; nt < 4; ++nt) {
      v16h bf;
#pragma unroll
      for (int h = 0; h < 16; ++h) {
        int kk = h + 16 * g;
        bf[h] = vT[kk * 64 + nt * 16 + m];
      }
      acc[nt] = __builtin_amdgcn_wmma_f32_16x16x32_f16(
          false, pa, false, bf, (short)0, acc[nt], false, false);
    }
    __syncthreads();
  }

#pragma unroll
  for (int nt = 0; nt < 4; ++nt)
#pragma unroll
    for (int i = 0; i < 8; ++i) {
      int row = q0 + i + 8 * g;
      int col = hh * HD_ + nt * 16 + m;
      float v = acc[nt][i] / lrow[i];
      Att[((size_t)(b * S_) + row) * D_ + col] = (_Float16)v;
    }
}

// ---------------------------------------------------------------------------
// LayerNorm (in place) then multiply by pooling weight mask/masksum
// ---------------------------------------------------------------------------
__global__ __launch_bounds__(256)
void ln_kernel(float* __restrict__ Hbuf, const float* __restrict__ gamma,
               const float* __restrict__ beta, const int* __restrict__ mask,
               const float* __restrict__ msum) {
  __shared__ float red[256];
  int r = blockIdx.x;      // token row in [0, BS)
  int d = threadIdx.x;
  int b = r / S_;
  float x = Hbuf[(size_t)r * D_ + d];
  red[d] = x; __syncthreads();
  for (int s = 128; s > 0; s >>= 1) { if (d < s) red[d] += red[d + s]; __syncthreads(); }
  float mean = red[0] * (1.0f / D_);
  __syncthreads();
  float diff = x - mean;
  red[d] = diff * diff; __syncthreads();
  for (int s = 128; s > 0; s >>= 1) { if (d < s) red[d] += red[d + s]; __syncthreads(); }
  float var = red[0] * (1.0f / D_);
  float y = diff * rsqrtf(var + EPS_) * gamma[d] + beta[d];
  float w = (float)mask[r] / msum[b];
  Hbuf[(size_t)r * D_ + d] = y * w;
}

// ---------------------------------------------------------------------------
// Per-batch pooled sum + final 1-dim head (deterministic, no atomics)
// ---------------------------------------------------------------------------
__global__ __launch_bounds__(256)
void pool_logits_kernel(const float* __restrict__ Hbuf, const float* __restrict__ Wh,
                        const float* __restrict__ bh, float* __restrict__ out) {
  __shared__ float red[256];
  int b = blockIdx.x, d = threadIdx.x;
  float s = 0.0f;
  for (int ss = 0; ss < S_; ++ss) s += Hbuf[((size_t)b * S_ + ss) * D_ + d];
  red[d] = s * Wh[d]; __syncthreads();
  for (int k = 128; k > 0; k >>= 1) { if (d < k) red[d] += red[d + k]; __syncthreads(); }
  if (d == 0) out[b] = red[0] + bh[0];
}

// ---------------------------------------------------------------------------
extern "C" void kernel_launch(void* const* d_in, const int* in_sizes, int n_in,
                              void* d_out, int out_size, void* d_ws, size_t ws_size,
                              hipStream_t stream) {
  const int*   ids   = (const int*)d_in[0];
  const int*   amask = (const int*)d_in[1];
  const float* embT  = (const float*)d_in[2];
  const float* Wq    = (const float*)d_in[3];
  const float* Wk    = (const float*)d_in[4];
  const float* Wv    = (const float*)d_in[5];
  const float* Wo    = (const float*)d_in[6];
  const float* bo    = (const float*)d_in[7];
  const float* gamma = (const float*)d_in[8];
  const float* beta  = (const float*)d_in[9];
  const float* Wh    = (const float*)d_in[10];
  const float* bh    = (const float*)d_in[11];
  float* out = (float*)d_out;

  char* ws = (char*)d_ws;
  const size_t ACT = (size_t)BS_ * D_ * sizeof(_Float16);   // 8 MB
  _Float16* wsE   = (_Float16*)(ws);
  _Float16* wsQ   = (_Float16*)(ws + 1 * ACT);
  _Float16* wsK   = (_Float16*)(ws + 2 * ACT);
  _Float16* wsV   = (_Float16*)(ws + 3 * ACT);
  _Float16* wsAtt = (_Float16*)(ws + 4 * ACT);
  _Float16* wsW16 = (_Float16*)(ws + 5 * ACT);              // 4 * 64K halves
  float*    wsH   = (float*)   (ws + 5 * ACT + 4 * 65536 * sizeof(_Float16));
  float*    wsMs  = (float*)   (ws + 5 * ACT + 4 * 65536 * sizeof(_Float16)
                                   + (size_t)BS_ * D_ * sizeof(float));

  // prep
  convert_w_kernel<<<(4 * 65536) / 256, 256, 0, stream>>>(Wq, Wk, Wv, Wo, wsW16);
  gather_kernel<<<(BS_ * D_) / 256, 256, 0, stream>>>(ids, embT, wsE);
  masksum_kernel<<<B_, 256, 0, stream>>>(amask, wsMs);

  // QKV projections (Q pre-scaled by 1/sqrt(64) = 0.125)
  dim3 ggrid(BS_ / 64, D_ / 64);
  gemm128<0><<<ggrid, 128, 0, stream>>>(wsE, wsW16 + 0 * 65536, wsQ, nullptr,
                                        0.125f, nullptr, nullptr, nullptr);
  gemm128<0><<<ggrid, 128, 0, stream>>>(wsE, wsW16 + 1 * 65536, wsK, nullptr,
                                        1.0f, nullptr, nullptr, nullptr);
  gemm128<0><<<ggrid, 128, 0, stream>>>(wsE, wsW16 + 2 * 65536, wsV, nullptr,
                                        1.0f, nullptr, nullptr, nullptr);

  // flash attention
  dim3 agrid(S_ / 16, H_, B_);
  attention_kernel<<<agrid, 32, 0, stream>>>(wsQ, wsK, wsV, amask, wsAtt);

  // output projection + residual (re-gather emb in epilogue) + bias
  gemm128<1><<<ggrid, 128, 0, stream>>>(wsAtt, wsW16 + 3 * 65536, nullptr, wsH,
                                        1.0f, ids, embT, bo);

  // layernorm * pooling weight, then pooled head
  ln_kernel<<<BS_, 256, 0, stream>>>(wsH, gamma, beta, amask, wsMs);
  pool_logits_kernel<<<B_, 256, 0, stream>>>(wsH, Wh, bh, out);
}
